// NCFG_61632780697647
// MI455X (gfx1250) — matrix operand order repaired and checked by reading, not compiled, not run on hardware
//
#include <hip/hip_runtime.h>
#include <hip/hip_bf16.h>
#include <math.h>

// ---------------------------------------------------------------------------
// NCFG forward for MI455X (gfx1250, wave32, WMMA + async global->LDS).
//   DIM=64, B=4096, K=64, L=2. One block (128 thr = 4 wave32) per batch b.
// ---------------------------------------------------------------------------

typedef __attribute__((ext_vector_type(16))) _Float16 v16h;
typedef __attribute__((ext_vector_type(8)))  _Float16 v8h;
typedef __attribute__((ext_vector_type(8)))  float    v8f;

// GCC-style vector to match the async-to-LDS builtin's parameter type (V4i)
typedef int v4i_t __attribute__((vector_size(16)));
typedef __attribute__((address_space(1))) v4i_t* g_v4i_p;   // global int4*
typedef __attribute__((address_space(3))) v4i_t* l_v4i_p;   // LDS int4*

// Explicit global-address-space weight pointer (keeps loads as global_load_*)
typedef __attribute__((address_space(1))) const _Float16 gchalf;
typedef __attribute__((address_space(1))) const v16h    gv16h;

#define NB 4096
#define NK 64
#define ND 64
#define LROWF 68   // f32 LDS row stride (272B, 16B aligned, bank-friendly)
#define LROWH 72   // f16 LDS row stride (144B, 16B aligned)

#define LOG2E  1.4426950408889634f
#define LOG2E2 2.8853900817779268f

#if __has_builtin(__builtin_amdgcn_global_load_async_to_lds_b128) && \
    __has_builtin(__builtin_amdgcn_s_wait_asynccnt)
#define HAVE_ASYNC_LDS 1
#endif

__device__ __forceinline__ float fast_rcp(float x)  { return __builtin_amdgcn_rcpf(x); }
__device__ __forceinline__ float fast_exp(float x)  { return __builtin_amdgcn_exp2f(x * LOG2E); }
__device__ __forceinline__ float fast_sigmoid(float x) {
  return fast_rcp(1.0f + __builtin_amdgcn_exp2f(-x * LOG2E));
}
__device__ __forceinline__ float fast_tanh(float x) {
  float a = fabsf(x);
  float e = __builtin_amdgcn_exp2f(a * LOG2E2);      // exp(2|x|)
  float r = 1.0f - 2.0f * fast_rcp(e + 1.0f);        // tanh(|x|), inf-safe
  return copysignf(r, x);
}

// 16B global -> LDS copy (async on gfx1250 when available)
__device__ __forceinline__ void cp16(const float* g, float* l) {
#ifdef HAVE_ASYNC_LDS
  __builtin_amdgcn_global_load_async_to_lds_b128(
      (g_v4i_p)(unsigned long long)g,
      (l_v4i_p)(unsigned int)(unsigned long long)l,
      0, 0);
#else
  float4 v = *(const float4*)g;
  *(float4*)l = v;
#endif
}
__device__ __forceinline__ void gather_fence() {
#ifdef HAVE_ASYNC_LDS
  __builtin_amdgcn_s_wait_asynccnt(0);
#endif
}

// A-fragment (16x32 f16) from an f32 LDS tile, per ISA A layout:
//   lanes 0-15 : a[0..7]=K+0..7,  a[8..15]=K+16..23   (row = M)
//   lanes 16-31: a[0..7]=K+8..15, a[8..15]=K+24..31
__device__ __forceinline__ v16h load_a_frag_f32(const float* tile, int row, int kstep, int lhalf) {
  const float* p = tile + row * LROWF + kstep * 32 + lhalf * 8;
  float4 a0 = *(const float4*)(p);
  float4 a1 = *(const float4*)(p + 4);
  float4 b0 = *(const float4*)(p + 16);
  float4 b1 = *(const float4*)(p + 20);
  v16h a;
  a[0]  = (_Float16)a0.x; a[1]  = (_Float16)a0.y; a[2]  = (_Float16)a0.z; a[3]  = (_Float16)a0.w;
  a[4]  = (_Float16)a1.x; a[5]  = (_Float16)a1.y; a[6]  = (_Float16)a1.z; a[7]  = (_Float16)a1.w;
  a[8]  = (_Float16)b0.x; a[9]  = (_Float16)b0.y; a[10] = (_Float16)b0.z; a[11] = (_Float16)b0.w;
  a[12] = (_Float16)b1.x; a[13] = (_Float16)b1.y; a[14] = (_Float16)b1.z; a[15] = (_Float16)b1.w;
  return a;
}

// Same fragment from an f16 LDS tile (h1)
__device__ __forceinline__ v16h load_a_frag_f16(const _Float16* tile, int row, int kstep, int lhalf) {
  const _Float16* p = tile + row * LROWH + kstep * 32 + lhalf * 8;
  v8h lo = *(const v8h*)(p);
  v8h hi = *(const v8h*)(p + 16);
  v16h a;
#pragma unroll
  for (int i = 0; i < 8; ++i) { a[i] = lo[i]; a[8 + i] = hi[i]; }
  return a;
}

// B-fragment (32x16 f16): B[k][n] = W[n][k]; W row-major f16 -> two 16B chunks
// (k = 8*lhalf+0..7 and 16+8*lhalf+0..7, matching the A layout).
// Addrspace(1) pointer => global_load_b128 (LOADcnt only, no flat/DScnt path).
// Cross-layer hoisting is prevented by laundering the pointer per iteration.
__device__ __forceinline__ v16h load_b_frag(gchalf* W, int ncol, int kstep, int lhalf) {
  typedef __attribute__((address_space(1))) const v8h gv8h;
  gchalf* p = W + ncol * 64 + kstep * 32 + lhalf * 8;
  v8h lo = *(gv8h*)(p);
  v8h hi = *(gv8h*)(p + 16);
  v16h r;
#pragma unroll
  for (int i = 0; i < 8; ++i) { r[i] = lo[i]; r[8 + i] = hi[i]; }
  return r;
}

__device__ __forceinline__ void mm_a32(const float* Atile, gchalf* W,
                                       int mrow, int l16, int lhalf, v8f acc[4]) {
#pragma unroll
  for (int ks = 0; ks < 2; ++ks) {
    v16h a = load_a_frag_f32(Atile, mrow, ks, lhalf);
#pragma unroll
    for (int nt = 0; nt < 4; ++nt) {
      v16h bf = load_b_frag(W, nt * 16 + l16, ks, lhalf);
      acc[nt] = __builtin_amdgcn_wmma_f32_16x16x32_f16(
          false, a, false, bf, (short)0, acc[nt], false, false);
    }
  }
}

__device__ __forceinline__ void mm_a16(const _Float16* Atile, gchalf* W,
                                       int mrow, int l16, int lhalf, v8f acc[4]) {
#pragma unroll
  for (int ks = 0; ks < 2; ++ks) {
    v16h a = load_a_frag_f16(Atile, mrow, ks, lhalf);
#pragma unroll
    for (int nt = 0; nt < 4; ++nt) {
      v16h bf = load_b_frag(W, nt * 16 + l16, ks, lhalf);
      acc[nt] = __builtin_amdgcn_wmma_f32_16x16x32_f16(
          false, a, false, bf, (short)0, acc[nt], false, false);
    }
  }
}

// dot of n floats (n % 4 == 0), a in LDS, w in global, both 16B aligned
__device__ __forceinline__ float dotf4(const float* a, const float* w, int n) {
  float s = 0.0f;
  for (int j = 0; j < n; j += 4) {
    float4 x = *(const float4*)(a + j);
    float4 y = *(const float4*)(w + j);
    s += x.x * y.x + x.y * y.y + x.z * y.z + x.w * y.w;
  }
  return s;
}

// ---------------------------------------------------------------------------
// Prep: f32 -> f16 weights, fused RNN bias.  Workspace:
//   [0..8191] W_ih f16 | [8192..16383] W_hh f16 | [16384..16639] bias1 f32[64]
// ---------------------------------------------------------------------------
__global__ void ncfg_prep(const float* __restrict__ Wih, const float* __restrict__ Whh,
                          const float* __restrict__ bih, const float* __restrict__ bhh,
                          _Float16* __restrict__ Wih16, _Float16* __restrict__ Whh16,
                          float* __restrict__ bias1) {
  int t = blockIdx.x * blockDim.x + threadIdx.x;
  if (t < ND * ND) {
    Wih16[t] = (_Float16)Wih[t];
    Whh16[t] = (_Float16)Whh[t];
  }
  if (t < ND) bias1[t] = bih[t] + bhh[t];
}

// ---------------------------------------------------------------------------
// Main: one block per batch element; each wave owns 16 neighbor rows.
// ---------------------------------------------------------------------------
__global__ __launch_bounds__(128, 4) void ncfg_main(
    const int* __restrict__ items, const int* __restrict__ heads,
    const int* __restrict__ rels,  const int* __restrict__ tails,
    const float* __restrict__ emb, const float* __restrict__ trans,
    const float* __restrict__ attn_w, const float* __restrict__ attn_b,
    const _Float16* __restrict__ Wih16, const _Float16* __restrict__ Whh16,
    const float* __restrict__ bias1,
    const float* __restrict__ l1_w, const float* __restrict__ l1_b,
    const float* __restrict__ l2_w, const float* __restrict__ l2_b,
    const float* __restrict__ l3_w, const float* __restrict__ l3_b,
    const float* __restrict__ lin_w, const float* __restrict__ lin_b,
    float* __restrict__ out) {
  __shared__ __align__(16) float    sh_h[NK * LROWF];   // head rows (f32)
  __shared__ __align__(16) float    sh_t[NK * LROWF];   // tail rows (f32)
  __shared__ __align__(16) float    sh_r[NK * LROWF];   // relation rows (f32)
  __shared__ __align__(16) _Float16 sh_h1[NK * LROWH];  // h1 (f16)
  __shared__ __align__(16) float    s_arr[NK];
  __shared__ __align__(16) float    o_acc[ND];
  __shared__ __align__(16) float    item_v[ND];
  __shared__ __align__(16) float    ybuf[2][ND];

  const int b     = blockIdx.x;
  const int tid   = threadIdx.x;
  const int wave  = tid >> 5;
  const int lane  = tid & 31;
  const int lhalf = lane >> 4;
  const int l16   = lane & 15;
  const int mbase = wave * 16;    // this wave's neighbor-row base
  const int mrow  = mbase + l16;

  if (tid < ND) {
    o_acc[tid] = 0.0f;
    item_v[tid] = emb[(size_t)items[b] * ND + tid];
  }

  for (int l = 0; l < 2; ++l) {
    // Launder weight pointers each iteration (as addrspace(1) so loads stay
    // global_load_b128): blocks cross-layer pinning of 128 VGPRs of weight
    // fragments; within an iteration they are ordinary cached loads.
    gchalf* wih = (gchalf*)(unsigned long long)Wih16;
    gchalf* whh = (gchalf*)(unsigned long long)Whh16;
    asm volatile("" : "+s"(wih), "+s"(whh));

    // ---- wave-local gather: lane pair handles one row, half-row each ------
    {
      const int ib  = (l * NB + b) * NK;
      const int row = mbase + (lane >> 1);
      const int seg = (lane & 1) * 32;           // 32 floats per half-row
      const int hidx = heads[ib + row];
      const int tidx = tails[ib + row];
      const int ridx = rels[ib + row];
      const float* gh = emb   + (size_t)hidx * ND + seg;
      const float* gt = emb   + (size_t)tidx * ND + seg;
      const float* gr = trans + (size_t)ridx * ND + seg;
      float* lh = sh_h + row * LROWF + seg;
      float* lt = sh_t + row * LROWF + seg;
      float* lr = sh_r + row * LROWF + seg;
#pragma unroll
      for (int i = 0; i < 8; ++i) cp16(gh + i * 4, lh + i * 4);
#pragma unroll
      for (int i = 0; i < 8; ++i) cp16(gt + i * 4, lt + i * 4);
#pragma unroll
      for (int i = 0; i < 8; ++i) cp16(gr + i * 4, lr + i * 4);
      gather_fence();   // ASYNCcnt==0: rows visible to this wave
    }

    // ---- attention logit for row mbase+l16 (lane pair splits the 192-dot) -
    {
      const float* hr = sh_h + mrow * LROWF;
      const float* rr = sh_r + mrow * LROWF;
      const float* tr = sh_t + mrow * LROWF;
      float p;
      if (lhalf == 0)
        p = dotf4(hr, attn_w, 64) + dotf4(rr, attn_w + 64, 32);
      else
        p = dotf4(rr + 32, attn_w + 96, 32) + dotf4(tr, attn_w + 128, 64);
      p += __shfl_xor(p, 16, 32);
      if (lhalf == 0) s_arr[mrow] = fast_sigmoid(p + attn_b[0]);
    }
    __syncthreads();   // s_arr complete across waves

    // ---- matmul 1: h @ W_ih^T -> h1 = tanh(. + bias1) -> LDS f16 ----------
    v8f acc1[4];
#pragma unroll
    for (int nt = 0; nt < 4; ++nt)
#pragma unroll
      for (int i = 0; i < 8; ++i) acc1[nt][i] = 0.0f;
    mm_a32(sh_h, wih, mrow, l16, lhalf, acc1);

#pragma unroll
    for (int nt = 0; nt < 4; ++nt) {
      int col = nt * 16 + l16;
      float bb = bias1[col];
#pragma unroll
      for (int r = 0; r < 8; ++r) {
        int row = mbase + r + lhalf * 8;
        sh_h1[row * LROWH + col] = (_Float16)fast_tanh(acc1[nt][r] + bb);
      }
    }

    // ---- matmul 2+3 fused into one accumulator: tl@W_ih^T + h1@W_hh^T -----
    v8f acc[4];
#pragma unroll
    for (int nt = 0; nt < 4; ++nt)
#pragma unroll
      for (int i = 0; i < 8; ++i) acc[nt][i] = 0.0f;
    mm_a32(sh_t,  wih, mrow, l16, lhalf, acc);
    mm_a16(sh_h1, whh, mrow, l16, lhalf, acc);  // same-wave LDS RAW: in-order

    // ---- softmax (wave-level shuffle reduction) + weighted h2 sum ---------
    {
      float v0 = s_arr[lane];
      float v1 = s_arr[lane + 32];
      float m = fmaxf(v0, v1);
#pragma unroll
      for (int o = 16; o > 0; o >>= 1) m = fmaxf(m, __shfl_xor(m, o, 32));
      float e = fast_exp(v0 - m) + fast_exp(v1 - m);
#pragma unroll
      for (int o = 16; o > 0; o >>= 1) e += __shfl_xor(e, o, 32);
      float inv = fast_rcp(e);

#pragma unroll
      for (int nt = 0; nt < 4; ++nt) {
        int col = nt * 16 + l16;
        float bb = bias1[col];
        float part = 0.0f;
#pragma unroll
        for (int r = 0; r < 8; ++r) {
          int row  = mbase + r + lhalf * 8;
          float pi = fast_exp(s_arr[row] - m) * inv;
          part += pi * fast_tanh(acc[nt][r] + bb);
        }
        atomicAdd(&o_acc[col], part);
      }
    }
    __syncthreads();   // s_arr/o_acc settled before LDS reuse or head
  }

  // ---- NCF head (64 threads, float4-vectorized) ---------------------------
  if (tid < ND) {
    float a = l1_b[tid];
    const float4* w4 = (const float4*)(l1_w + tid * 128);
    const float4* u4 = (const float4*)o_acc;
    const float4* i4 = (const float4*)item_v;
    for (int j = 0; j < 16; ++j) {
      float4 w = w4[j], u = u4[j];
      a += w.x * u.x + w.y * u.y + w.z * u.z + w.w * u.w;
    }
    for (int j = 0; j < 16; ++j) {
      float4 w = w4[16 + j], v = i4[j];
      a += w.x * v.x + w.y * v.y + w.z * v.z + w.w * v.w;
    }
    ybuf[0][tid] = fmaxf(a, 0.0f);
  }
  __syncthreads();
  if (tid < ND) {
    float a = l2_b[tid];
    const float4* w4 = (const float4*)(l2_w + tid * 64);
    const float4* y4 = (const float4*)ybuf[0];
    for (int j = 0; j < 16; ++j) {
      float4 w = w4[j], y = y4[j];
      a += w.x * y.x + w.y * y.y + w.z * y.z + w.w * y.w;
    }
    ybuf[1][tid] = fmaxf(a, 0.0f);
  }
  __syncthreads();
  if (tid < ND) {
    float a = l3_b[tid];
    const float4* w4 = (const float4*)(l3_w + tid * 64);
    const float4* y4 = (const float4*)ybuf[1];
    for (int j = 0; j < 16; ++j) {
      float4 w = w4[j], y = y4[j];
      a += w.x * y.x + w.y * y.y + w.z * y.z + w.w * y.w;
    }
    ybuf[0][tid] = fast_sigmoid(a);
  }
  __syncthreads();
  if (tid == 0) {
    float s = lin_b[0];
    const float4* y4 = (const float4*)ybuf[0];
    const float4* u4 = (const float4*)o_acc;
    const float4* i4 = (const float4*)item_v;
    const float4* wa = (const float4*)lin_w;
    const float4* wb = (const float4*)(lin_w + 64);
    for (int j = 0; j < 16; ++j) {
      float4 y = y4[j], w = wa[j];
      s += y.x * w.x + y.y * w.y + y.z * w.z + y.w * w.w;
      float4 u = u4[j], v = i4[j], w2 = wb[j];
      s += u.x * v.x * w2.x + u.y * v.y * w2.y + u.z * v.z * w2.z + u.w * v.w * w2.w;
    }
    out[b] = fast_sigmoid(s);
  }
}

// ---------------------------------------------------------------------------
extern "C" void kernel_launch(void* const* d_in, const int* in_sizes, int n_in,
                              void* d_out, int out_size, void* d_ws, size_t ws_size,
                              hipStream_t stream) {
  const int*   items  = (const int*)d_in[0];
  const int*   heads  = (const int*)d_in[1];
  const int*   rels   = (const int*)d_in[2];
  const int*   tails  = (const int*)d_in[3];
  const float* emb    = (const float*)d_in[4];
  const float* trans  = (const float*)d_in[5];
  const float* attn_w = (const float*)d_in[6];
  const float* attn_b = (const float*)d_in[7];
  const float* W_ih   = (const float*)d_in[8];
  const float* W_hh   = (const float*)d_in[9];
  const float* b_ih   = (const float*)d_in[10];
  const float* b_hh   = (const float*)d_in[11];
  const float* l1_w   = (const float*)d_in[12];
  const float* l1_b   = (const float*)d_in[13];
  const float* l2_w   = (const float*)d_in[14];
  const float* l2_b   = (const float*)d_in[15];
  const float* l3_w   = (const float*)d_in[16];
  const float* l3_b   = (const float*)d_in[17];
  const float* lin_w  = (const float*)d_in[18];
  const float* lin_b  = (const float*)d_in[19];

  _Float16* Wih16 = (_Float16*)d_ws;
  _Float16* Whh16 = (_Float16*)((char*)d_ws + 8192);
  float*    bias1 = (float*)((char*)d_ws + 16384);

  ncfg_prep<<<16, 256, 0, stream>>>(W_ih, W_hh, b_ih, b_hh, Wih16, Whh16, bias1);

  ncfg_main<<<NB, 128, 0, stream>>>(
      items, heads, rels, tails, emb, trans, attn_w, attn_b,
      Wih16, Whh16, bias1,
      l1_w, l1_b, l2_w, l2_b, l3_w, l3_b, lin_w, lin_b,
      (float*)d_out);
}
